// MoETransformerBlock_61091614819117
// MI455X (gfx1250) — matrix operand order repaired and checked by reading, not compiled, not run on hardware
//
#include <hip/hip_runtime.h>
#include <hip/hip_bf16.h>
#include <math.h>

#define SEQ        4096
#define STATE_DIM  512
#define ACTION_DIM 496
#define T_DIM      16
#define EMBED      1024
#define HEADS      4
#define HEAD_DIM   256
#define FF         4096
#define N_EXP      8
#define LN_EPS     1e-5f

typedef _Float16 f16;
typedef __attribute__((ext_vector_type(16))) _Float16 v16h;
typedef __attribute__((ext_vector_type(8)))  _Float16 v8h;
typedef __attribute__((ext_vector_type(8)))  float    v8f;

// ---------------------------------------------------------------------------
// WMMA fragment helpers (layouts per CDNA5 ISA, wave32).
// Pointers are per-lane bases hoisted out of the k-loops so the inner loop is
// pure {global_load_b128 + v_wmma} with immediate offsets.
// ---------------------------------------------------------------------------

// A 16x32 f16: lane base p = &A[row + (lane&15)][hi*8]; element e ->
// k = (e<8 ? e : e+8) + hi*8
__device__ __forceinline__ v16h a_frag(const f16* __restrict__ p, int k0) {
  v8h c0 = *(const v8h*)(p + k0);
  v8h c1 = *(const v8h*)(p + k0 + 16);
  v16h a;
#pragma unroll
  for (int i = 0; i < 8; ++i) { a[i] = c0[i]; a[8 + i] = c1[i]; }
  return a;
}

// B 32x16 f16 from B^T [N,K] row-major: lane base p = &Bt[n][hi*8];
// element e -> k = k0 + (e<8 ? e : e+8) + hi*8
__device__ __forceinline__ v16h b_frag(const f16* __restrict__ p, int k0) {
  v8h c0 = *(const v8h*)(p + k0);
  v8h c1 = *(const v8h*)(p + k0 + 16);
  v16h b;
#pragma unroll
  for (int i = 0; i < 8; ++i) { b[i] = c0[i]; b[8 + i] = c1[i]; }
  return b;
}

__device__ __forceinline__ v8f wmma_f16(v16h a, v16h b, v8f c) {
  return __builtin_amdgcn_wmma_f32_16x16x32_f16(false, a, false, b, (short)0, c, false, false);
}

// ---------------------------------------------------------------------------
// 1) h = concat(x, time_emb, state); also f16 copy
// ---------------------------------------------------------------------------
__global__ __launch_bounds__(256)
void build_h_kernel(const float* __restrict__ x, const int* __restrict__ t,
                    const float* __restrict__ st, const float* __restrict__ tt,
                    float* __restrict__ h32, f16* __restrict__ h16) {
  size_t idx = (size_t)blockIdx.x * 256 + threadIdx.x;
  int s = (int)(idx / EMBED), c = (int)(idx % EMBED);
  float v;
  if (c < ACTION_DIM)               v = x[(size_t)s * ACTION_DIM + c];
  else if (c < ACTION_DIM + T_DIM)  v = tt[t[s] * T_DIM + (c - ACTION_DIM)];
  else                              v = st[(size_t)s * STATE_DIM + (c - ACTION_DIM - T_DIM)];
  h32[idx] = v;
  h16[idx] = (f16)v;
}

// ---------------------------------------------------------------------------
// 2) gate logits + softmax + top-2 -> comb[S, N_EXP] (one wave32 per row)
// ---------------------------------------------------------------------------
__global__ __launch_bounds__(256)
void gate_topk_kernel(const float* __restrict__ h32, const float* __restrict__ gw,
                      const float* __restrict__ gb, float* __restrict__ comb) {
  int wave = threadIdx.x >> 5, lane = threadIdx.x & 31;
  int s = blockIdx.x * 8 + wave;
  float acc[N_EXP];
#pragma unroll
  for (int e = 0; e < N_EXP; ++e) acc[e] = 0.f;
  const float* hr = h32 + (size_t)s * EMBED;
  for (int k = lane; k < EMBED; k += 32) {
    float hv = hr[k];
    const float* g = gw + (size_t)k * N_EXP;
#pragma unroll
    for (int e = 0; e < N_EXP; ++e) acc[e] += hv * g[e];
  }
#pragma unroll
  for (int e = 0; e < N_EXP; ++e)
#pragma unroll
    for (int d = 1; d < 32; d <<= 1) acc[e] += __shfl_xor(acc[e], d, 32);
  float p[N_EXP];
  float mx = -3.0e38f;
#pragma unroll
  for (int e = 0; e < N_EXP; ++e) { acc[e] += gb[e]; mx = fmaxf(mx, acc[e]); }
  float den = 0.f;
#pragma unroll
  for (int e = 0; e < N_EXP; ++e) { p[e] = __expf(acc[e] - mx); den += p[e]; }
  float inv = 1.f / den;
#pragma unroll
  for (int e = 0; e < N_EXP; ++e) p[e] *= inv;
  int i1 = 0;
#pragma unroll
  for (int e = 1; e < N_EXP; ++e) if (p[e] > p[i1]) i1 = e;
  int i2 = (i1 == 0) ? 1 : 0;
#pragma unroll
  for (int e = 0; e < N_EXP; ++e) if (e != i1 && p[e] > p[i2]) i2 = e;
  if (lane == 0) {
#pragma unroll
    for (int e = 0; e < N_EXP; ++e) comb[(size_t)s * N_EXP + e] = 0.f;
    comb[(size_t)s * N_EXP + i1] = p[i1];
    comb[(size_t)s * N_EXP + i2] = p[i2];
  }
}

// ---------------------------------------------------------------------------
// 3) weight convert: f32 [K,N] -> f16 [N,K] via LDS 32x32 tile
//    (coalesced loads AND coalesced f16 stores)
// ---------------------------------------------------------------------------
__global__ __launch_bounds__(256)
void convT_kernel(const float* __restrict__ W, f16* __restrict__ Wt, int K, int N) {
  __shared__ f16 tile[32][33];
  size_t base = (size_t)blockIdx.z * K * N;
  int kb = blockIdx.x * 32, nb = blockIdx.y * 32;
  int tc = threadIdx.x & 31, tr = threadIdx.x >> 5;
#pragma unroll
  for (int i = 0; i < 4; ++i) {
    int k = kb + tr + i * 8, n = nb + tc;
    f16 v = (f16)0.f;
    if (k < K && n < N) v = (f16)W[base + (size_t)k * N + n];
    tile[tr + i * 8][tc] = v;
  }
  __syncthreads();
#pragma unroll
  for (int i = 0; i < 4; ++i) {
    int n = nb + tr + i * 8, k = kb + tc;
    if (n < N && k < K) Wt[base + (size_t)n * K + k] = tile[tc][tr + i * 8];
  }
}

// ---------------------------------------------------------------------------
// 4) generic WMMA GEMM: C[M,N] = act(A[M,K] @ Bt^T + bias)
//    block = 256 thr (8 waves), tile 64x256, wave tile 32x64 (2x4 WMMA)
//    Column index CLAMPED (not guarded) -> no exec-mask ops in the k-loop;
//    out-of-range columns compute garbage that is never stored.
// ---------------------------------------------------------------------------
__global__ __launch_bounds__(256)
void gemm_f16_kernel(const f16* __restrict__ A, const f16* __restrict__ Bt,
                     const float* __restrict__ bias, float* __restrict__ C32,
                     f16* __restrict__ C16, int M, int N, int K, int relu) {
  int wave = threadIdx.x >> 5, lane = threadIdx.x & 31;
  int m16 = lane & 15, hi = lane >> 4;
  int row0 = blockIdx.x * 64 + (wave >> 2) * 32;
  int col0 = blockIdx.y * 256 + (wave & 3) * 64;

  const f16* pa0 = A + (size_t)(row0 + m16) * K + hi * 8;
  const f16* pa1 = pa0 + (size_t)16 * K;
  const f16* pb[4];
#pragma unroll
  for (int j = 0; j < 4; ++j) {
    int n = col0 + j * 16 + m16;
    int nc = n < N ? n : N - 1;            // clamp, never stored when OOB
    pb[j] = Bt + (size_t)nc * K + hi * 8;
  }

  v8f acc[2][4];
#pragma unroll
  for (int i = 0; i < 2; ++i)
#pragma unroll
    for (int j = 0; j < 4; ++j) acc[i][j] = {};

#pragma unroll 2
  for (int k0 = 0; k0 < K; k0 += 32) {
    __builtin_prefetch(pa0 + k0 + 256, 0, 0);   // global_prefetch_b8
    __builtin_prefetch(pb[0] + k0 + 256, 0, 0);
    v16h a0 = a_frag(pa0, k0);
    v16h a1 = a_frag(pa1, k0);
#pragma unroll
    for (int j = 0; j < 4; ++j) {
      v16h b = b_frag(pb[j], k0);
      acc[0][j] = wmma_f16(a0, b, acc[0][j]);
      acc[1][j] = wmma_f16(a1, b, acc[1][j]);
    }
  }

#pragma unroll
  for (int i = 0; i < 2; ++i)
#pragma unroll
    for (int j = 0; j < 4; ++j) {
      int n = col0 + j * 16 + m16;
      if (n >= N) continue;
      float bv = bias ? bias[n] : 0.f;
#pragma unroll
      for (int r = 0; r < 8; ++r) {
        int m = row0 + i * 16 + hi * 8 + r;
        float v = acc[i][j][r] + bv;
        if (relu) v = fmaxf(v, 0.f);
        if (C32) C32[(size_t)m * N + n] = v;
        if (C16) C16[(size_t)m * N + n] = (f16)v;
      }
    }
}

// ---------------------------------------------------------------------------
// 5) MoE GEMM: moe[s,:] = sum_e comb[s,e] * (h[s,:] @ W_e + b_e)
//    A rows scaled by comb (exact); expert skipped when no row in the wave
//    routes to it (__ballot).
// ---------------------------------------------------------------------------
__global__ __launch_bounds__(256)
void moe_gemm_kernel(const f16* __restrict__ H, const f16* __restrict__ Wt,
                     const float* __restrict__ comb, const float* __restrict__ eb,
                     float* __restrict__ moe32, f16* __restrict__ moe16) {
  int wave = threadIdx.x >> 5, lane = threadIdx.x & 31;
  int m16 = lane & 15, hi = lane >> 4;
  int row0 = blockIdx.x * 64 + (wave >> 2) * 32;
  int col0 = blockIdx.y * 256 + (wave & 3) * 64;

  const f16* pa0 = H + (size_t)(row0 + m16) * EMBED + hi * 8;
  const f16* pa1 = pa0 + (size_t)16 * EMBED;

  v8f acc[2][4];
#pragma unroll
  for (int i = 0; i < 2; ++i)
#pragma unroll
    for (int j = 0; j < 4; ++j) acc[i][j] = {};

  for (int e = 0; e < N_EXP; ++e) {
    float sc0 = comb[(size_t)(row0 + m16) * N_EXP + e];
    float sc1 = comb[(size_t)(row0 + 16 + m16) * N_EXP + e];
    if (__ballot(sc0 != 0.f || sc1 != 0.f) == 0ull) continue;
    const f16* pw = Wt + (size_t)e * EMBED * EMBED + (size_t)(col0 + m16) * EMBED + hi * 8;
    f16 h0 = (f16)sc0, h1 = (f16)sc1;
#pragma unroll 2
    for (int k0 = 0; k0 < EMBED; k0 += 32) {
      v16h a0 = a_frag(pa0, k0);
      v16h a1 = a_frag(pa1, k0);
#pragma unroll
      for (int i = 0; i < 16; ++i) { a0[i] = a0[i] * h0; a1[i] = a1[i] * h1; }
#pragma unroll
      for (int j = 0; j < 4; ++j) {
        v16h b = b_frag(pw + (size_t)j * 16 * EMBED, k0);
        acc[0][j] = wmma_f16(a0, b, acc[0][j]);
        acc[1][j] = wmma_f16(a1, b, acc[1][j]);
      }
    }
  }

#pragma unroll
  for (int i = 0; i < 2; ++i)
#pragma unroll
    for (int j = 0; j < 4; ++j) {
      int n = col0 + j * 16 + m16;
#pragma unroll
      for (int r = 0; r < 8; ++r) {
        int m = row0 + i * 16 + hi * 8 + r;
        float v = acc[i][j][r];
#pragma unroll
        for (int e = 0; e < N_EXP; ++e)
          v += comb[(size_t)m * N_EXP + e] * eb[(size_t)e * EMBED + n];
        moe32[(size_t)m * EMBED + n] = v;
        moe16[(size_t)m * EMBED + n] = (f16)v;
      }
    }
}

// ---------------------------------------------------------------------------
// 6) attn prep: scale q by 1/sqrt(HEAD_DIM)=1/16 in place; V -> [h][d][s]
// ---------------------------------------------------------------------------
__global__ __launch_bounds__(256)
void attn_prep_kernel(f16* __restrict__ qkv, f16* __restrict__ vT) {
  size_t idx = (size_t)blockIdx.x * 256 + threadIdx.x;
  int s = (int)(idx / EMBED), c = (int)(idx % EMBED);
  f16* qp = qkv + (size_t)s * 3 * EMBED + c;
  *qp = (f16)((float)*qp * 0.0625f);
  f16 v = qkv[(size_t)s * 3 * EMBED + 2 * EMBED + c];
  int hh = c / HEAD_DIM, d = c % HEAD_DIM;
  vT[((size_t)hh * HEAD_DIM + d) * SEQ + s] = v;
}

// ---------------------------------------------------------------------------
// 7) flash attention: one wave32 = 16 query rows of one head.
//    Per 32-key chunk: 16 WMMA (Q.K^T) + online softmax (16-lane shfl
//    reductions matching the D-tile row layout) + P transpose via per-wave
//    LDS tile + 16 WMMA (P.V). All global pointers are per-lane bases
//    advanced by constant strides.
// ---------------------------------------------------------------------------
__global__ __launch_bounds__(256)
void flash_attn_kernel(const f16* __restrict__ qkv, const f16* __restrict__ vT,
                       f16* __restrict__ ao) {
  __shared__ f16 pbuf[8][16 * 32];
  int wave = threadIdx.x >> 5, lane = threadIdx.x & 31;
  int m16 = lane & 15, hi = lane >> 4;
  int h = blockIdx.y;
  int q0 = (blockIdx.x * 8 + wave) * 16;

  // Q fragments cached in registers (rows q0..q0+15, full d=256)
  const f16* pq = qkv + (size_t)(q0 + m16) * 3 * EMBED + h * HEAD_DIM + hi * 8;
  v16h qa[8];
#pragma unroll
  for (int c = 0; c < 8; ++c) qa[c] = a_frag(pq, c * 32);

  // per-lane K base: column n = key row m16; advances by 32 keys per chunk
  const f16* pk = qkv + (size_t)m16 * 3 * EMBED + EMBED + h * HEAD_DIM + hi * 8;
  // per-lane V^T base: column n = d index m16; k = key; advances by 32 keys
  const f16* pv = vT + (size_t)h * HEAD_DIM * SEQ + (size_t)m16 * SEQ + hi * 8;

  v8f o[16];
#pragma unroll
  for (int dt = 0; dt < 16; ++dt) o[dt] = {};
  float mrow[8], lrow[8];
#pragma unroll
  for (int r = 0; r < 8; ++r) { mrow[r] = -3.0e38f; lrow[r] = 0.f; }
  f16* myp = pbuf[wave];

  for (int kb = 0; kb < SEQ; kb += 32) {
    v8f s0 = {}, s1 = {};
#pragma unroll
    for (int c = 0; c < 8; ++c) {
      v16h b0 = b_frag(pk, c * 32);                       // keys kb..kb+15
      v16h b1 = b_frag(pk + (size_t)16 * 3 * EMBED, c * 32); // keys kb+16..31
      s0 = wmma_f16(qa[c], b0, s0);
      s1 = wmma_f16(qa[c], b1, s1);
    }
    float scale[8];
#pragma unroll
    for (int r = 0; r < 8; ++r) {
      float mx = fmaxf(s0[r], s1[r]);
#pragma unroll
      for (int d = 1; d < 16; d <<= 1) mx = fmaxf(mx, __shfl_xor(mx, d, 32));
      float nm = fmaxf(mrow[r], mx);
      scale[r] = __expf(mrow[r] - nm);
      mrow[r] = nm;
      float p0 = __expf(s0[r] - nm);
      float p1 = __expf(s1[r] - nm);
      float ps = p0 + p1;
#pragma unroll
      for (int d = 1; d < 16; d <<= 1) ps += __shfl_xor(ps, d, 32);
      lrow[r] = lrow[r] * scale[r] + ps;
      int m = hi * 8 + r;
      myp[m * 32 + m16]      = (f16)p0;
      myp[m * 32 + 16 + m16] = (f16)p1;
    }
    __syncthreads();                 // wave's DS stores -> DS loads ordering
    v16h pa = a_frag(myp + (size_t)m16 * 32 + hi * 8, 0);
#pragma unroll
    for (int dt = 0; dt < 16; ++dt) {
      v8f t = o[dt];
#pragma unroll
      for (int r = 0; r < 8; ++r) t[r] *= scale[r];
      v16h vb = b_frag(pv + (size_t)dt * 16 * SEQ, kb);
      o[dt] = wmma_f16(pa, vb, t);
    }
    pk += (size_t)32 * 3 * EMBED;
  }

  int nn = m16;
#pragma unroll
  for (int dt = 0; dt < 16; ++dt)
#pragma unroll
    for (int r = 0; r < 8; ++r) {
      float v = o[dt][r] / lrow[r];
      ao[(size_t)(q0 + hi * 8 + r) * EMBED + h * HEAD_DIM + dt * 16 + nn] = (f16)v;
    }
}

// ---------------------------------------------------------------------------
// 8) residual add + LayerNorm (one block per row)
// ---------------------------------------------------------------------------
__global__ __launch_bounds__(256)
void add_ln_kernel(const float* __restrict__ A, const float* __restrict__ B,
                   const float* __restrict__ g, const float* __restrict__ bb,
                   float* __restrict__ out32, f16* __restrict__ out16, int N) {
  __shared__ float s1[256], s2[256];
  int row = blockIdx.x, tid = threadIdx.x;
  float vals[4];
  float sum = 0.f, sq = 0.f;
  int cnt = 0;
  for (int i = tid; i < N; i += 256) {
    float v = A[(size_t)row * N + i] + (B ? B[(size_t)row * N + i] : 0.f);
    vals[cnt++] = v;
    sum += v; sq += v * v;
  }
  s1[tid] = sum; s2[tid] = sq;
  __syncthreads();
  for (int off = 128; off > 0; off >>= 1) {
    if (tid < off) { s1[tid] += s1[tid + off]; s2[tid] += s2[tid + off]; }
    __syncthreads();
  }
  float mean = s1[0] / N;
  float var  = s2[0] / N - mean * mean;
  float rs = rsqrtf(var + LN_EPS);
  cnt = 0;
  for (int i = tid; i < N; i += 256) {
    float o = (vals[cnt++] - mean) * rs * g[i] + bb[i];
    if (out32) out32[(size_t)row * N + i] = o;
    if (out16) out16[(size_t)row * N + i] = (f16)o;
  }
}

// ---------------------------------------------------------------------------
// workspace layout
// ---------------------------------------------------------------------------
static constexpr size_t OFF_H32   = 0;
static constexpr size_t OFF_H16   = OFF_H32   + (size_t)SEQ * EMBED * 4;
static constexpr size_t OFF_COMB  = OFF_H16   + (size_t)SEQ * EMBED * 2;
static constexpr size_t OFF_MOE32 = OFF_COMB  + (size_t)SEQ * N_EXP * 4;
static constexpr size_t OFF_MOE16 = OFF_MOE32 + (size_t)SEQ * EMBED * 4;
static constexpr size_t OFF_QKV16 = OFF_MOE16 + (size_t)SEQ * EMBED * 2;
static constexpr size_t OFF_VT16  = OFF_QKV16 + (size_t)SEQ * 3 * EMBED * 2;
static constexpr size_t OFF_AO16  = OFF_VT16  + (size_t)HEADS * HEAD_DIM * SEQ * 2;
static constexpr size_t OFF_PRJ32 = OFF_AO16  + (size_t)SEQ * EMBED * 2;
static constexpr size_t OFF_O116  = OFF_PRJ32 + (size_t)SEQ * EMBED * 4;
static constexpr size_t OFF_FF116 = OFF_O116  + (size_t)SEQ * EMBED * 2;
static constexpr size_t OFF_FF232 = OFF_FF116 + (size_t)SEQ * FF * 2;
static constexpr size_t OFF_EXPT  = OFF_FF232 + (size_t)SEQ * ACTION_DIM * 4;
static constexpr size_t OFF_IPT   = OFF_EXPT  + (size_t)N_EXP * EMBED * EMBED * 2;
static constexpr size_t OFF_OPT   = OFF_IPT   + (size_t)EMBED * 3 * EMBED * 2;
static constexpr size_t OFF_F1T   = OFF_OPT   + (size_t)EMBED * EMBED * 2;
static constexpr size_t OFF_F2T   = OFF_F1T   + (size_t)EMBED * FF * 2;

static inline int ceil_div(int a, int b) { return (a + b - 1) / b; }

extern "C" void kernel_launch(void* const* d_in, const int* in_sizes, int n_in,
                              void* d_out, int out_size, void* d_ws, size_t ws_size,
                              hipStream_t stream) {
  const float* x         = (const float*)d_in[0];
  const int*   timei     = (const int*)  d_in[1];
  const float* state     = (const float*)d_in[2];
  const float* ttab      = (const float*)d_in[3];
  const float* expert_w  = (const float*)d_in[4];
  const float* expert_b  = (const float*)d_in[5];
  const float* gate_w    = (const float*)d_in[6];
  const float* gate_b    = (const float*)d_in[7];
  const float* in_proj_w = (const float*)d_in[8];
  const float* in_proj_b = (const float*)d_in[9];
  const float* out_proj_w= (const float*)d_in[10];
  const float* out_proj_b= (const float*)d_in[11];
  const float* ln1_g     = (const float*)d_in[12];
  const float* ln1_b     = (const float*)d_in[13];
  const float* ff1_w     = (const float*)d_in[14];
  const float* ff1_b     = (const float*)d_in[15];
  const float* ff2_w     = (const float*)d_in[16];
  const float* ff2_b     = (const float*)d_in[17];
  const float* ln2_g     = (const float*)d_in[18];
  const float* ln2_b     = (const float*)d_in[19];

  char* ws = (char*)d_ws;
  float* h32   = (float*)(ws + OFF_H32);
  f16*   h16   = (f16*)  (ws + OFF_H16);
  float* comb  = (float*)(ws + OFF_COMB);
  float* moe32 = (float*)(ws + OFF_MOE32);
  f16*   moe16 = (f16*)  (ws + OFF_MOE16);
  f16*   qkv16 = (f16*)  (ws + OFF_QKV16);
  f16*   vT16  = (f16*)  (ws + OFF_VT16);
  f16*   ao16  = (f16*)  (ws + OFF_AO16);
  float* prj32 = (float*)(ws + OFF_PRJ32);
  f16*   o116  = (f16*)  (ws + OFF_O116);
  f16*   ff116 = (f16*)  (ws + OFF_FF116);
  float* ff232 = (float*)(ws + OFF_FF232);
  f16*   expT  = (f16*)  (ws + OFF_EXPT);
  f16*   ipT   = (f16*)  (ws + OFF_IPT);
  f16*   opT   = (f16*)  (ws + OFF_OPT);
  f16*   f1T   = (f16*)  (ws + OFF_F1T);
  f16*   f2T   = (f16*)  (ws + OFF_F2T);

  // 1. h and gate
  build_h_kernel<<<(SEQ * EMBED) / 256, 256, 0, stream>>>(x, timei, state, ttab, h32, h16);
  gate_topk_kernel<<<SEQ / 8, 256, 0, stream>>>(h32, gate_w, gate_b, comb);

  // 2. weight transposes to f16 [N,K] (LDS tiled)
  convT_kernel<<<dim3(EMBED / 32, EMBED / 32, N_EXP), 256, 0, stream>>>(expert_w, expT, EMBED, EMBED);
  convT_kernel<<<dim3(EMBED / 32, (3 * EMBED) / 32, 1), 256, 0, stream>>>(in_proj_w, ipT, EMBED, 3 * EMBED);
  convT_kernel<<<dim3(EMBED / 32, EMBED / 32, 1), 256, 0, stream>>>(out_proj_w, opT, EMBED, EMBED);
  convT_kernel<<<dim3(EMBED / 32, FF / 32, 1), 256, 0, stream>>>(ff1_w, f1T, EMBED, FF);
  convT_kernel<<<dim3(FF / 32, ceil_div(ACTION_DIM, 32), 1), 256, 0, stream>>>(ff2_w, f2T, FF, ACTION_DIM);

  // 3. MoE
  moe_gemm_kernel<<<dim3(SEQ / 64, EMBED / 256), 256, 0, stream>>>(h16, expT, comb, expert_b, moe32, moe16);

  // 4. QKV projection
  gemm_f16_kernel<<<dim3(SEQ / 64, (3 * EMBED) / 256), 256, 0, stream>>>(
      moe16, ipT, in_proj_b, nullptr, qkv16, SEQ, 3 * EMBED, EMBED, 0);

  // 5. attention
  attn_prep_kernel<<<(SEQ * EMBED) / 256, 256, 0, stream>>>(qkv16, vT16);
  flash_attn_kernel<<<dim3(SEQ / 128, HEADS), 256, 0, stream>>>(qkv16, vT16, ao16);

  // 6. out_proj
  gemm_f16_kernel<<<dim3(SEQ / 64, EMBED / 256), 256, 0, stream>>>(
      ao16, opT, out_proj_b, prj32, nullptr, SEQ, EMBED, EMBED, 0);

  // 7. LN1(moe + attn)
  add_ln_kernel<<<SEQ, 256, 0, stream>>>(moe32, prj32, ln1_g, ln1_b, nullptr, o116, EMBED);

  // 8. FF
  gemm_f16_kernel<<<dim3(SEQ / 64, FF / 256), 256, 0, stream>>>(
      o116, f1T, ff1_b, nullptr, ff116, SEQ, FF, EMBED, 1);
  gemm_f16_kernel<<<dim3(SEQ / 64, ceil_div(ACTION_DIM, 256)), 256, 0, stream>>>(
      ff116, f2T, ff2_b, ff232, nullptr, SEQ, ACTION_DIM, FF, 0);

  // 9. LN2 -> d_out (f32)
  add_ln_kernel<<<SEQ, 256, 0, stream>>>(ff232, nullptr, ln2_g, ln2_b, (float*)d_out, nullptr, ACTION_DIM);
}
